// CNNMRF_59528246722987
// MI455X (gfx1250) — matrix operand order repaired and checked by reading, not compiled, not run on hardware
//
#include <hip/hip_runtime.h>

typedef __attribute__((ext_vector_type(16))) _Float16 v16h;
typedef __attribute__((ext_vector_type(8)))  _Float16 v8h;
typedef __attribute__((ext_vector_type(8)))  float    v8f;

// ---------------------------------------------------------------------------
// Deterministic block reduction (256 threads = 8 waves of 32)
// ---------------------------------------------------------------------------
__device__ __forceinline__ float block_reduce_sum(float v) {
    __shared__ float s[8];
    #pragma unroll
    for (int off = 16; off > 0; off >>= 1) v += __shfl_xor(v, off, 32);
    const int lane = threadIdx.x & 31;
    const int w    = threadIdx.x >> 5;
    if (lane == 0) s[w] = v;
    __syncthreads();
    v = (threadIdx.x < 8) ? s[threadIdx.x] : 0.0f;
    if (w == 0) {
        #pragma unroll
        for (int off = 4; off > 0; off >>= 1) v += __shfl_xor(v, off, 32);
    }
    return v; // valid in thread 0
}

// ---------------------------------------------------------------------------
// Patch extraction: feat [C,H,W] -> q [Qp, D] f16 row-major (zero padded rows)
// d = c*9 + kh*3 + kw ; patch i = (pi, pj) row-major over (Ho, Wo), stride 2
// grid: (D/256, Qp)
// ---------------------------------------------------------------------------
__global__ __launch_bounds__(256) void pack_q(const float* __restrict__ feat,
                                              _Float16* __restrict__ qf16,
                                              int C, int H, int W, int Wo,
                                              int Q, int D) {
    const int d = blockIdx.x * 256 + threadIdx.x;
    const int i = blockIdx.y;
    if (d >= D) return;
    float v = 0.0f;
    if (i < Q) {
        const int pi = i / Wo, pj = i - pi * Wo;
        const int c = d / 9, rm = d - c * 9;
        const int kh = rm / 3, kw = rm - kh * 3;
        v = feat[((size_t)c * H + (pi * 2 + kh)) * W + (pj * 2 + kw)];
    }
    qf16[(size_t)i * D + d] = (_Float16)v;
}

// ---------------------------------------------------------------------------
// Style patches: f32 [P,D] -> f16 [Pp,D] (zero padded) + inv_norm (f32 sums)
// grid: Pp blocks, one block per row
// ---------------------------------------------------------------------------
__global__ __launch_bounds__(256) void pack_sp(const float* __restrict__ sp,
                                               _Float16* __restrict__ spf16,
                                               float* __restrict__ invn,
                                               int P, int D) {
    const int p = blockIdx.x;
    float ss = 0.0f;
    for (int d = threadIdx.x; d < D; d += 256) {
        float v = (p < P) ? sp[(size_t)p * D + d] : 0.0f;
        spf16[(size_t)p * D + d] = (_Float16)v;
        ss += v * v;
    }
    ss = block_reduce_sum(ss);
    if (threadIdx.x == 0) invn[p] = (p < P) ? rsqrtf(ss) : 0.0f;
}

// ---------------------------------------------------------------------------
// Fused correlation GEMM + scaled argmax.
// Block = 256 threads (8 waves), owns 32 q-rows (2 M-tiles). Each wave owns
// chunks of 4 adjacent 16-wide p-tiles and accumulates a 2x4 grid of 16x16
// f32 tiles with v_wmma_f32_16x16x32_f16 (B fragments reused across both
// M-tiles -> ~24 B/lane per WMMA of L2 traffic). Epilogue keeps a per-row
// running argmax of resp[q,p] * inv_norm[p] (first-max tie-break).
// ---------------------------------------------------------------------------
__global__ __launch_bounds__(256) void mrf_gemm_argmax(
        const _Float16* __restrict__ qf16,   // [Qp, D], Qp % 32 == 0
        const _Float16* __restrict__ spf16,  // [Pp, D], Pp == Ptiles*16
        const float*    __restrict__ invn,   // [Pp]
        int*            __restrict__ amax,   // [Qp]
        int Q, int D, int Ptiles, int P) {
    const int wave = threadIdx.x >> 5;
    const int lane = threadIdx.x & 31;
    const int g    = lane >> 4;   // half-wave group
    const int n    = lane & 15;   // column-in-tile (B/C) and row for A loads
    const int q0   = blockIdx.x * 32;

    // A: 16x32 f16 fragment. Lane (g, n) holds row m=n, K = {8g..8g+7, 16+8g..16+8g+7}
    const _Float16* arow0 = qf16 + (size_t)(q0 + n) * D;        // M-tile 0
    const _Float16* arow1 = arow0 + (size_t)16 * D;             // M-tile 1

    float best[2][8];
    int   bidx[2][8];
    #pragma unroll
    for (int mt = 0; mt < 2; ++mt)
        #pragma unroll
        for (int r = 0; r < 8; ++r) { best[mt][r] = -INFINITY; bidx[mt][r] = 0x7fffffff; }

    for (int chunk = wave * 4; chunk < Ptiles; chunk += 32) {
        const _Float16* brow[4];
        int  pcol[4];
        bool tvalid[4];
        #pragma unroll
        for (int t = 0; t < 4; ++t) {
            const int pt = chunk + t;
            tvalid[t] = (pt < Ptiles);
            pcol[t]   = tvalid[t] ? (pt * 16 + n) : n;      // safe row when invalid
            // B: 32x16 f16 fragment. Lane (g, n) holds col N=n, K = {8g..8g+7, 16+8g..16+8g+7},
            // i.e. two runs of 8 halves of sp row pcol at offsets k+8g and k+16+8g.
            brow[t] = spf16 + (size_t)pcol[t] * D;
        }

        v8f acc[2][4];
        const v8f vzero = {};
        #pragma unroll
        for (int mt = 0; mt < 2; ++mt)
            #pragma unroll
            for (int t = 0; t < 4; ++t) acc[mt][t] = vzero;

        for (int k = 0; k < D; k += 32) {
            v8h a0lo = *(const v8h*)(arow0 + k + 8 * g);
            v8h a0hi = *(const v8h*)(arow0 + k + 16 + 8 * g);
            v16h a0 = __builtin_shufflevector(a0lo, a0hi,
                      0,1,2,3,4,5,6,7,8,9,10,11,12,13,14,15);
            v8h a1lo = *(const v8h*)(arow1 + k + 8 * g);
            v8h a1hi = *(const v8h*)(arow1 + k + 16 + 8 * g);
            v16h a1 = __builtin_shufflevector(a1lo, a1hi,
                      0,1,2,3,4,5,6,7,8,9,10,11,12,13,14,15);
            #pragma unroll
            for (int t = 0; t < 4; ++t) {
                v8h blo = *(const v8h*)(brow[t] + k + 8 * g);
                v8h bhi = *(const v8h*)(brow[t] + k + 16 + 8 * g);
                v16h b = __builtin_shufflevector(blo, bhi,
                         0,1,2,3,4,5,6,7,8,9,10,11,12,13,14,15);
                acc[0][t] = __builtin_amdgcn_wmma_f32_16x16x32_f16(
                    false, a0, false, b, (short)0, acc[0][t], false, false);
                acc[1][t] = __builtin_amdgcn_wmma_f32_16x16x32_f16(
                    false, a1, false, b, (short)0, acc[1][t], false, false);
            }
        }

        // Epilogue: scale by inv_norm and fold into running argmax.
        // C layout: VGPR r, lanes 0-15 -> M=r ; lanes 16-31 -> M=8+r  => m = r + 8g
        #pragma unroll
        for (int t = 0; t < 4; ++t) {
            if (!tvalid[t]) continue;            // uniform per wave
            const bool  pv = (pcol[t] < P);
            const float in = pv ? invn[pcol[t]] : 0.0f;
            #pragma unroll
            for (int mt = 0; mt < 2; ++mt)
                #pragma unroll
                for (int r = 0; r < 8; ++r) {
                    const float s = acc[mt][t][r] * in;
                    if (pv && (s > best[mt][r] ||
                               (s == best[mt][r] && pcol[t] < bidx[mt][r]))) {
                        best[mt][r] = s; bidx[mt][r] = pcol[t];
                    }
                }
        }
    }

    // Reduce across the 16 lanes of each half-wave (same rows, different cols).
    #pragma unroll
    for (int mt = 0; mt < 2; ++mt)
        #pragma unroll
        for (int r = 0; r < 8; ++r) {
            #pragma unroll
            for (int off = 1; off < 16; off <<= 1) {
                const float ob = __shfl_xor(best[mt][r], off, 32);
                const int   oi = __shfl_xor(bidx[mt][r], off, 32);
                if (ob > best[mt][r] || (ob == best[mt][r] && oi < bidx[mt][r])) {
                    best[mt][r] = ob; bidx[mt][r] = oi;
                }
            }
        }

    __shared__ float sb[8][32];
    __shared__ int   si[8][32];
    if (n == 0) {
        #pragma unroll
        for (int mt = 0; mt < 2; ++mt)
            #pragma unroll
            for (int r = 0; r < 8; ++r) {
                sb[wave][mt * 16 + r + 8 * g] = best[mt][r];
                si[wave][mt * 16 + r + 8 * g] = bidx[mt][r];
            }
    }
    __syncthreads();
    if (threadIdx.x < 32) {
        const int m = threadIdx.x;
        float bb = sb[0][m]; int bi = si[0][m];
        #pragma unroll
        for (int w2 = 1; w2 < 8; ++w2) {
            const float ob = sb[w2][m]; const int oi = si[w2][m];
            if (ob > bb || (ob == bb && oi < bi)) { bb = ob; bi = oi; }
        }
        if (q0 + m < Q) amax[q0 + m] = bi;
    }
}

// ---------------------------------------------------------------------------
// Full-precision matched-patch difference:  partial[i] = sum_d (q[i,d]-sp[idx,d])^2
// grid: Q blocks
// ---------------------------------------------------------------------------
__global__ __launch_bounds__(256) void mrf_diff(const float* __restrict__ feat,
                                                const float* __restrict__ spf32,
                                                const int*   __restrict__ amax,
                                                float* __restrict__ partial,
                                                int C, int H, int W, int Wo, int D) {
    const int i   = blockIdx.x;
    const int idx = amax[i];
    const int pi = i / Wo, pj = i - pi * Wo;
    float ss = 0.0f;
    for (int d = threadIdx.x; d < D; d += 256) {
        const int c = d / 9, rm = d - c * 9;
        const int kh = rm / 3, kw = rm - kh * 3;
        const float qv = feat[((size_t)c * H + (pi * 2 + kh)) * W + (pj * 2 + kw)];
        const float sv = spf32[(size_t)idx * D + d];
        const float df = qv - sv;
        ss += df * df;
    }
    ss = block_reduce_sum(ss);
    if (threadIdx.x == 0) partial[i] = ss;
}

// ---------------------------------------------------------------------------
// Content loss partials: sum (a-b)^2
// ---------------------------------------------------------------------------
__global__ __launch_bounds__(256) void sq_diff(const float* __restrict__ a,
                                               const float* __restrict__ b,
                                               float* __restrict__ partial, int N) {
    float ss = 0.0f;
    for (int i = blockIdx.x * 256 + threadIdx.x; i < N; i += gridDim.x * 256) {
        const float d = a[i] - b[i];
        ss += d * d;
    }
    ss = block_reduce_sum(ss);
    if (threadIdx.x == 0) partial[blockIdx.x] = ss;
}

// ---------------------------------------------------------------------------
// TV loss partials on [1,3,512,512]; shift cancels in differences, scale doesn't.
// ---------------------------------------------------------------------------
__global__ __launch_bounds__(256) void tv_kernel(const float* __restrict__ img,
                                                 float* __restrict__ partial) {
    const int H = 512, W = 512, HW = H * W, N = 3 * HW;
    float ss = 0.0f;
    for (int id = blockIdx.x * 256 + threadIdx.x; id < N; id += gridDim.x * 256) {
        const int c = id >> 18;            // / (512*512)
        const int rm = id & (HW - 1);
        const int h = rm >> 9;             // / 512
        const int w = rm & 511;
        const float sc = (c == 0) ? (1.0f / 0.229f)
                       : (c == 1) ? (1.0f / 0.224f) : (1.0f / 0.225f);
        const int h1 = (h + 1 < H) ? h + 1 : h;
        const int w1 = (w + 1 < W) ? w + 1 : w;
        const float v0 = img[c * HW + h * W + w];
        const float gx = (img[c * HW + h1 * W + w] - v0) * sc;
        const float gy = (img[c * HW + h * W + w1] - v0) * sc;
        ss += gx * gx + gy * gy;
    }
    ss = block_reduce_sum(ss);
    if (threadIdx.x == 0) partial[blockIdx.x] = ss;
}

// ---------------------------------------------------------------------------
// Deterministic final combine.
// ---------------------------------------------------------------------------
__global__ __launch_bounds__(256) void final_combine(
        const float* __restrict__ pm3, int n3, float s3,
        const float* __restrict__ pm4, int n4, float s4,
        const float* __restrict__ pc,  int nc, float sc,
        const float* __restrict__ pt,  int nt, float st,
        float* __restrict__ out) {
    float acc = 0.0f;
    for (int i = threadIdx.x; i < n3; i += 256) acc += pm3[i] * s3;
    for (int i = threadIdx.x; i < n4; i += 256) acc += pm4[i] * s4;
    for (int i = threadIdx.x; i < nc; i += 256) acc += pc[i]  * sc;
    for (int i = threadIdx.x; i < nt; i += 256) acc += pt[i]  * st;
    acc = block_reduce_sum(acc);
    if (threadIdx.x == 0) out[0] = acc;
}

// ---------------------------------------------------------------------------
// Host launcher
// ---------------------------------------------------------------------------
extern "C" void kernel_launch(void* const* d_in, const int* in_sizes, int n_in,
                              void* d_out, int out_size, void* d_ws, size_t ws_size,
                              hipStream_t stream) {
    const float* synthesis = (const float*)d_in[0]; // [1,3,512,512]
    const float* feat3     = (const float*)d_in[1]; // [1,256,128,128]
    const float* feat4     = (const float*)d_in[2]; // [1,512,64,64]
    const float* feat42    = (const float*)d_in[3]; // [1,512,64,64]
    const float* sp3       = (const float*)d_in[4]; // [3969,256,3,3]
    const float* sp4       = (const float*)d_in[5]; // [961,512,3,3]
    const float* content   = (const float*)d_in[6]; // [1,512,64,64]
    float* out = (float*)d_out;

    // Level 3: C=256, H=W=128, Ho=Wo=63; Q padded to 32 rows, P to 16 cols.
    const int Q3 = 3969, Qp3 = 4000, D3 = 2304, Pt3 = 249, Pp3 = Pt3 * 16;
    // Level 4: C=512, H=W=64, Ho=Wo=31
    const int Q4 = 961,  Qp4 = 992,  D4 = 4608, Pt4 = 61,  Pp4 = Pt4 * 16;

    char* base = (char*)d_ws;
    size_t off = 0;
    auto take = [&](size_t bytes) -> void* {
        void* p = base + off;
        off += (bytes + 255) & ~(size_t)255;
        return p;
    };
    _Float16* q3f = (_Float16*)take((size_t)Qp3 * D3 * 2);
    _Float16* s3f = (_Float16*)take((size_t)Pp3 * D3 * 2);
    _Float16* q4f = (_Float16*)take((size_t)Qp4 * D4 * 2);
    _Float16* s4f = (_Float16*)take((size_t)Pp4 * D4 * 2);
    float* inv3 = (float*)take((size_t)Pp3 * 4);
    float* inv4 = (float*)take((size_t)Pp4 * 4);
    int*   idx3 = (int*)take((size_t)Qp3 * 4);
    int*   idx4 = (int*)take((size_t)Qp4 * 4);
    float* pm3  = (float*)take((size_t)Q3 * 4);
    float* pm4  = (float*)take((size_t)Q4 * 4);
    float* pc   = (float*)take((size_t)1024 * 4);
    float* pt   = (float*)take((size_t)512 * 4);
    (void)ws_size; (void)in_sizes; (void)n_in; (void)out_size;

    // Pack operands to f16 (+ style norms)
    pack_q<<<dim3(D3 / 256, Qp3), 256, 0, stream>>>(feat3, q3f, 256, 128, 128, 63, Q3, D3);
    pack_q<<<dim3(D4 / 256, Qp4), 256, 0, stream>>>(feat4, q4f, 512,  64,  64, 31, Q4, D4);
    pack_sp<<<Pp3, 256, 0, stream>>>(sp3, s3f, inv3, Q3, D3);
    pack_sp<<<Pp4, 256, 0, stream>>>(sp4, s4f, inv4, Q4, D4);

    // WMMA correlation + argmax (32 q-rows per block)
    mrf_gemm_argmax<<<Qp3 / 32, 256, 0, stream>>>(q3f, s3f, inv3, idx3, Q3, D3, Pt3, Q3);
    mrf_gemm_argmax<<<Qp4 / 32, 256, 0, stream>>>(q4f, s4f, inv4, idx4, Q4, D4, Pt4, Q4);

    // Full-precision loss terms
    mrf_diff<<<Q3, 256, 0, stream>>>(feat3, sp3, idx3, pm3, 256, 128, 128, 63, D3);
    mrf_diff<<<Q4, 256, 0, stream>>>(feat4, sp4, idx4, pm4, 512,  64,  64, 31, D4);
    sq_diff<<<1024, 256, 0, stream>>>(feat42, content, pc, 512 * 64 * 64);
    tv_kernel<<<512, 256, 0, stream>>>(synthesis, pt);

    final_combine<<<1, 256, 0, stream>>>(
        pm3, Q3, 1.0f / ((float)Q3 * (float)D3),
        pm4, Q4, 1.0f / ((float)Q4 * (float)D4),
        pc, 1024, 1.0f / (float)(512 * 64 * 64),
        pt, 512, 0.001f / (float)(3 * 512 * 512),
        out);
}
